// ImprovedSSM_22445499088920
// MI455X (gfx1250) — compile-verified, matrix operand order checked by reading, not run on hardware
//
#include <hip/hip_runtime.h>
#include <hip/hip_bf16.h>

typedef __attribute__((ext_vector_type(16))) __bf16 v16bf;
typedef __attribute__((ext_vector_type(8)))  float  v8f;
typedef __attribute__((ext_vector_type(4)))  unsigned int u32x4;
typedef __attribute__((ext_vector_type(8)))  int  i32x8;
typedef __attribute__((ext_vector_type(4)))  int  i32x4;

#define B_TOT  128
#define T_LEN  2048
#define IN_D   64
#define ADIM   16
#define S_DIM  256
#define H_DIM  512
#define NLAYER 2

#if defined(__has_builtin)
#if __has_builtin(__builtin_amdgcn_tensor_load_to_lds) && \
    __has_builtin(__builtin_amdgcn_s_wait_tensorcnt)
#define USE_TDM 1
#endif
#endif
#ifndef USE_TDM
#define USE_TDM 0
#endif

// bf16 workspace layout (element offsets)
#define WS_WP   0
#define WS_A    (WS_WP   + H_DIM * IN_D)
#define WS_BM   (WS_A    + NLAYER * S_DIM * S_DIM)
#define WS_CM   (WS_BM   + NLAYER * S_DIM * H_DIM)
#define WS_DM   (WS_CM   + NLAYER * H_DIM * S_DIM)
#define WS_PMW1 (WS_DM   + NLAYER * H_DIM * H_DIM)
#define WS_VW1  (WS_PMW1 + H_DIM * H_DIM)

__global__ void cvt_f32_bf16(const float* __restrict__ src, __bf16* __restrict__ dst, int n) {
  int i = blockIdx.x * blockDim.x + threadIdx.x;
  if (i < n) dst[i] = (__bf16)src[i];
}

// Load one 16x32 bf16 WMMA operand fragment (A from LDS, B from global) per the
// CDNA5 16-bit layout: lanes 0-15 hold K0..7 / K16..23, lanes 16-31 hold K8..15 / K24..31.
__device__ __forceinline__ v16bf load_frag(const __bf16* base, int stride, int kbase) {
  const int lane = threadIdx.x & 31;
  const int half = lane >> 4;
  const int m    = lane & 15;
  const __bf16* p = base + m * stride + kbase + half * 8;
  union { v16bf v; uint4 q[2]; } f;
  f.q[0] = *(const uint4*)(p);
  f.q[1] = *(const uint4*)(p + 16);
  return f.v;
}

template <int K>
__device__ __forceinline__ void gemm_acc(v8f& acc, const __bf16* aBase, int aStride,
                                         const __bf16* bBase, int bStride) {
#pragma unroll
  for (int k = 0; k < K; k += 32) {
    v16bf a = load_frag(aBase, aStride, k);
    v16bf b = load_frag(bBase, bStride, k);
    acc = __builtin_amdgcn_wmma_f32_16x16x32_bf16(false, a, false, b, (short)0, acc,
                                                  false, false);
  }
}

// C/D layout: VGPR v, lanes 0-15 -> row v; lanes 16-31 -> row v+8; col = ntile*16 + lane%16
__device__ __forceinline__ void store_tile_bf16(__bf16* base, int stride, int ntile,
                                                const v8f& acc) {
  const int lane = threadIdx.x & 31;
  const int half = lane >> 4;
  const int col  = ntile * 16 + (lane & 15);
#pragma unroll
  for (int v = 0; v < 8; ++v) base[(v + 8 * half) * stride + col] = (__bf16)acc[v];
}

__device__ __forceinline__ void ln_partial(const v8f& acc, int wv,
                                           float (*redS)[32], float (*redQ)[32]) {
  const int lane = threadIdx.x & 31;
  const int half = lane >> 4;
#pragma unroll
  for (int v = 0; v < 8; ++v) {
    float s = acc[v];
    float q = s * s;
#pragma unroll
    for (int m = 8; m >= 1; m >>= 1) {
      s += __shfl_xor(s, m, 16);
      q += __shfl_xor(q, m, 16);
    }
    if ((lane & 15) == 0) {
      const int row = v + 8 * half;
      redS[row][wv] = s;
      redQ[row][wv] = q;
    }
  }
}

__device__ __forceinline__ void ln_stats(int tid, float (*redS)[32], float (*redQ)[32]) {
  if (tid < 16) {
    float s = 0.f, q = 0.f;
#pragma unroll
    for (int w = 0; w < 32; ++w) { s += redS[tid][w]; q += redQ[tid][w]; }
    const float mean = s * (1.f / 512.f);
    const float var  = q * (1.f / 512.f) - mean * mean;
    redS[tid][0] = mean;
    redQ[tid][0] = rsqrtf(var + 1e-5f);
  }
}

#if USE_TDM
// Issue a TDM 2D tile load: 16 rows x 64 f32, row stride T_LEN*IN_D elements,
// from x[(b0..b0+15), t, :] into LDS at ldsDst. D# per CDNA5 ISA.
// This toolchain's builtin takes 6 args: (g0 u32x4, g1 i32x8, g2 i32x4, g3 i32x4,
// extra i32x8, cpol i32); groups 2/3/extra are unused for a 2D tile.
__device__ __forceinline__ void tdm_issue_x(const float* x, int b0, int t, float* ldsDst) {
  const unsigned long long ga =
      (unsigned long long)(uintptr_t)(x + ((size_t)b0 * T_LEN + t) * IN_D);
  const unsigned int lds = (unsigned int)(uintptr_t)ldsDst;  // low 32 bits = LDS offset
  u32x4 g0;
  g0[0] = 1u;                                   // count=1, user descriptor
  g0[1] = lds;                                  // lds_addr [63:32]
  g0[2] = (unsigned int)ga;                     // global_addr[31:0]
  g0[3] = (unsigned int)(ga >> 32) | (2u << 30);// global_addr[56:32] | type=2
  i32x8 g1;
  g1[0] = 0x00020000;                           // data_size=2 (4 bytes)
  g1[1] = (int)(IN_D << 16);                    // tensor_dim0[15:0] @ bits 63:48
  g1[2] = (int)(16 << 16);                      // tensor_dim1[15:0] @ bits 95:80
  g1[3] = (int)(IN_D << 16);                    // tile_dim0 @ bits 127:112
  g1[4] = 16;                                   // tile_dim1 @ bits 143:128
  g1[5] = (int)(T_LEN * IN_D);                  // tensor_dim0_stride[31:0]
  g1[6] = 0;
  g1[7] = 0;
  const i32x4 z4 = {0, 0, 0, 0};
  const i32x8 z8 = {0, 0, 0, 0, 0, 0, 0, 0};
  __builtin_amdgcn_tensor_load_to_lds(g0, g1, z4, z4, z8, 0);
}
#endif

__global__ __launch_bounds__(1024, 1)
void ssm_persistent(const float* __restrict__ x, const float* __restrict__ hidden0,
                    const float* __restrict__ bp,
                    const float* __restrict__ ln_g, const float* __restrict__ ln_b,
                    const float* __restrict__ pm_b1, const float* __restrict__ pm_g,
                    const float* __restrict__ pm_lb, const float* __restrict__ pm_W2,
                    const float* __restrict__ pm_b2, const float* __restrict__ log_std,
                    const float* __restrict__ v_b1, const float* __restrict__ v_g,
                    const float* __restrict__ v_lb, const float* __restrict__ v_W2,
                    const float* __restrict__ v_b2,
                    const __bf16* __restrict__ wbf, float* __restrict__ out) {
  __shared__ alignas(16) __bf16 sh_h[16 * H_DIM];    // activations (bf16)
  __shared__ alignas(16) __bf16 sh_sA[16 * S_DIM];   // hidden state ping
  __shared__ alignas(16) __bf16 sh_sB[16 * S_DIM];   // hidden state pong
  __shared__ alignas(16) __bf16 sh_t1[16 * H_DIM];   // post-tanh head activations
  __shared__ alignas(16) __bf16 sh_x[16 * IN_D];     // bf16 input tile
  __shared__ float redS[16][32];
  __shared__ float redQ[16][32];
#if USE_TDM
  __shared__ alignas(16) float sh_xf32[2][16 * IN_D];  // TDM double buffer
#endif

  const int tid  = threadIdx.x;
  const int wv   = tid >> 5;          // wave id == N-tile id
  const int lane = tid & 31;
  const int half = lane >> 4;
  const int col  = wv * 16 + (lane & 15);
  const int b0   = blockIdx.x * 16;   // batch-row tile

  const __bf16* Wp_bf  = wbf + WS_WP;
  const __bf16* A_bf   = wbf + WS_A;
  const __bf16* Bm_bf  = wbf + WS_BM;
  const __bf16* Cm_bf  = wbf + WS_CM;
  const __bf16* Dm_bf  = wbf + WS_DM;
  const __bf16* pW1_bf = wbf + WS_PMW1;
  const __bf16* vW1_bf = wbf + WS_VW1;

  float* out_means  = out;
  float* out_logstd = out + (size_t)B_TOT * T_LEN * ADIM;
  float* out_vals   = out + (size_t)2 * B_TOT * T_LEN * ADIM;
  float* out_hfin   = out_vals + (size_t)B_TOT * T_LEN;

  // Pin the loop-invariant Wp B-fragments in registers (round-1 codegen spilled
  // these to scratch after LICM hoisting).
  const __bf16* wpBase = Wp_bf + wv * 16 * IN_D;
  const v16bf wpb0 = load_frag(wpBase, IN_D, 0);
  const v16bf wpb1 = load_frag(wpBase, IN_D, 32);

  for (int i = tid; i < 16 * S_DIM; i += 1024)
    sh_sA[i] = (__bf16)hidden0[(size_t)(b0 + (i >> 8)) * S_DIM + (i & 255)];

#if USE_TDM
  if (wv == 0) tdm_issue_x(x, b0, 0, &sh_xf32[0][0]);
#endif
  __syncthreads();

  for (int t = 0; t < T_LEN; ++t) {
    // ---- stage x_t tile (16 x 64) as bf16 ----
#if USE_TDM
    if (wv == 0) __builtin_amdgcn_s_wait_tensorcnt(0);
    __syncthreads();
    {
      const int r = tid >> 6, c = tid & 63;
      sh_x[r * IN_D + c] = (__bf16)sh_xf32[t & 1][r * IN_D + c];
    }
    if (wv == 0 && (t + 1) < T_LEN)
      tdm_issue_x(x, b0, t + 1, &sh_xf32[(t + 1) & 1][0]);  // prefetch next step
#else
    {
      const int r = tid >> 6, c = tid & 63;
      sh_x[r * IN_D + c] = (__bf16)x[((size_t)(b0 + r) * T_LEN + t) * IN_D + c];
    }
#endif
    __syncthreads();

    // ---- h = relu(x @ Wp^T + bp) ----
    {
      v8f acc = {};
      v16bf a0 = load_frag(sh_x, IN_D, 0);
      v16bf a1 = load_frag(sh_x, IN_D, 32);
      acc = __builtin_amdgcn_wmma_f32_16x16x32_bf16(false, a0, false, wpb0, (short)0, acc,
                                                    false, false);
      acc = __builtin_amdgcn_wmma_f32_16x16x32_bf16(false, a1, false, wpb1, (short)0, acc,
                                                    false, false);
      const float bias = bp[col];
#pragma unroll
      for (int v = 0; v < 8; ++v) {
        float hv = fmaxf(acc[v] + bias, 0.f);
        sh_h[(v + 8 * half) * H_DIM + col] = (__bf16)hv;
      }
    }
    __syncthreads();

    // ---- SSM layers ----
#pragma unroll
    for (int i = 0; i < NLAYER; ++i) {
      const __bf16* stRd = (i == 0) ? sh_sA : sh_sB;
      __bf16*       stWr = (i == 0) ? sh_sB : sh_sA;
      v8f accO = {};
      if (wv < 16) {
        // next_state = hidden @ A^T + h @ Bm^T   (N=256 -> waves 0..15)
        v8f accS = {};
        gemm_acc<S_DIM>(accS, stRd, S_DIM, A_bf + i * S_DIM * S_DIM + wv * 16 * S_DIM, S_DIM);
        gemm_acc<H_DIM>(accS, sh_h, H_DIM, Bm_bf + i * S_DIM * H_DIM + wv * 16 * H_DIM, H_DIM);
        store_tile_bf16(stWr, S_DIM, wv, accS);
      } else {
        // start h @ Dm^T for N-tiles 16..31
        gemm_acc<H_DIM>(accO, sh_h, H_DIM, Dm_bf + i * H_DIM * H_DIM + wv * 16 * H_DIM, H_DIM);
      }
      __syncthreads();
      if (wv < 16)
        gemm_acc<H_DIM>(accO, sh_h, H_DIM, Dm_bf + i * H_DIM * H_DIM + wv * 16 * H_DIM, H_DIM);
      // + next_state @ Cm^T (all waves)
      gemm_acc<S_DIM>(accO, stWr, S_DIM, Cm_bf + i * H_DIM * S_DIM + wv * 16 * S_DIM, S_DIM);

      ln_partial(accO, wv, redS, redQ);
      __syncthreads();
      ln_stats(tid, redS, redQ);
      __syncthreads();
      const float g  = ln_g[i * H_DIM + col];
      const float bb = ln_b[i * H_DIM + col];
#pragma unroll
      for (int v = 0; v < 8; ++v) {
        const int row = v + 8 * half;
        float o = (accO[v] - redS[row][0]) * redQ[row][0] * g + bb;
        if (i > 0) o += (float)sh_h[row * H_DIM + col];
        sh_h[row * H_DIM + col] = (__bf16)o;
      }
      __syncthreads();
    }

    // ---- policy head: t1 = tanh(LN(h @ pm_W1^T + pm_b1)) ----
    v8f accP = {};
    gemm_acc<H_DIM>(accP, sh_h, H_DIM, pW1_bf + wv * 16 * H_DIM, H_DIM);
    {
      const float b1 = pm_b1[col];
#pragma unroll
      for (int v = 0; v < 8; ++v) accP[v] += b1;
    }
    ln_partial(accP, wv, redS, redQ);
    __syncthreads();
    ln_stats(tid, redS, redQ);
    __syncthreads();
    {
      const float g = pm_g[col], bb = pm_lb[col];
#pragma unroll
      for (int v = 0; v < 8; ++v) {
        const int row = v + 8 * half;
        float o = tanhf((accP[v] - redS[row][0]) * redQ[row][0] * g + bb);
        sh_t1[row * H_DIM + col] = (__bf16)o;
      }
    }
    __syncthreads();

    // ---- value head GEMM in registers (overlaps policy projection below) ----
    v8f accV = {};
    gemm_acc<H_DIM>(accV, sh_h, H_DIM, vW1_bf + wv * 16 * H_DIM, H_DIM);
    {
      const float b1 = v_b1[col];
#pragma unroll
      for (int v = 0; v < 8; ++v) accV[v] += b1;
    }
    // policy projection: mean = tanh(t1 @ pm_W2^T + pm_b2); also emit log_stds
    if (tid < 256) {
      const int r = tid >> 4, a = tid & 15;
      float dot = 0.f;
      for (int k = 0; k < H_DIM; ++k)
        dot += (float)sh_t1[r * H_DIM + k] * pm_W2[a * H_DIM + k];
      const size_t idx = ((size_t)(b0 + r) * T_LEN + t) * ADIM + a;
      out_means[idx]  = tanhf(dot + pm_b2[a]);
      out_logstd[idx] = log_std[a];
    }
    __syncthreads();
    ln_partial(accV, wv, redS, redQ);
    __syncthreads();
    ln_stats(tid, redS, redQ);
    __syncthreads();
    {
      const float g = v_g[col], bb = v_lb[col];
#pragma unroll
      for (int v = 0; v < 8; ++v) {
        const int row = v + 8 * half;
        float o = tanhf((accV[v] - redS[row][0]) * redQ[row][0] * g + bb);
        sh_t1[row * H_DIM + col] = (__bf16)o;
      }
    }
    __syncthreads();
    if (tid < 16) {
      float dot = 0.f;
      for (int k = 0; k < H_DIM; ++k)
        dot += (float)sh_t1[tid * H_DIM + k] * v_W2[k];
      out_vals[(size_t)(b0 + tid) * T_LEN + t] = dot + v_b2[0];
    }
    __syncthreads();
  }

  // hidden_final lives in sh_sA after layer 1 of the last step
  for (int i = tid; i < 16 * S_DIM; i += 1024)
    out_hfin[(size_t)(b0 + (i >> 8)) * S_DIM + (i & 255)] = (float)sh_sA[i];
}

extern "C" void kernel_launch(void* const* d_in, const int* in_sizes, int n_in,
                              void* d_out, int out_size, void* d_ws, size_t ws_size,
                              hipStream_t stream) {
  (void)in_sizes; (void)n_in; (void)out_size; (void)ws_size;
  const float* x       = (const float*)d_in[0];
  const float* hidden0 = (const float*)d_in[1];
  const float* Wp      = (const float*)d_in[2];
  const float* bp      = (const float*)d_in[3];
  const float* A       = (const float*)d_in[4];
  const float* Bm      = (const float*)d_in[5];
  const float* Cm      = (const float*)d_in[6];
  const float* Dm      = (const float*)d_in[7];
  const float* ln_g    = (const float*)d_in[8];
  const float* ln_b    = (const float*)d_in[9];
  const float* pm_W1   = (const float*)d_in[10];
  const float* pm_b1   = (const float*)d_in[11];
  const float* pm_g    = (const float*)d_in[12];
  const float* pm_lb   = (const float*)d_in[13];
  const float* pm_W2   = (const float*)d_in[14];
  const float* pm_b2   = (const float*)d_in[15];
  const float* log_std = (const float*)d_in[16];
  const float* v_W1    = (const float*)d_in[17];
  const float* v_b1    = (const float*)d_in[18];
  const float* v_g     = (const float*)d_in[19];
  const float* v_lb    = (const float*)d_in[20];
  const float* v_W2    = (const float*)d_in[21];
  const float* v_b2    = (const float*)d_in[22];

  __bf16* wbf = (__bf16*)d_ws;
  auto cvt = [&](const float* src, int off, int n) {
    cvt_f32_bf16<<<(n + 255) / 256, 256, 0, stream>>>(src, wbf + off, n);
  };
  cvt(Wp,    WS_WP,   H_DIM * IN_D);
  cvt(A,     WS_A,    NLAYER * S_DIM * S_DIM);
  cvt(Bm,    WS_BM,   NLAYER * S_DIM * H_DIM);
  cvt(Cm,    WS_CM,   NLAYER * H_DIM * S_DIM);
  cvt(Dm,    WS_DM,   NLAYER * H_DIM * H_DIM);
  cvt(pm_W1, WS_PMW1, H_DIM * H_DIM);
  cvt(v_W1,  WS_VW1,  H_DIM * H_DIM);

  ssm_persistent<<<B_TOT / 16, 1024, 0, stream>>>(
      x, hidden0, bp, ln_g, ln_b, pm_b1, pm_g, pm_lb, pm_W2, pm_b2, log_std,
      v_b1, v_g, v_lb, v_W2, v_b2, wbf, (float*)d_out);
}
